// CapsuleLayer_38122129719402
// MI455X (gfx1250) — matrix operand order repaired and checked by reading, not compiled, not run on hardware
//
#include <hip/hip_runtime.h>
#include <math.h>

typedef __attribute__((ext_vector_type(16))) _Float16 v16h;
typedef __attribute__((ext_vector_type(8)))  float    v8f;

#define NCAPS  64
#define NVOTES 36
#define DIN    256
#define DH     128
#define DCP    32

// Output region offsets (flat floats, reference tuple order)
#define OFF_VOTES 0UL
#define OFF_SCALE 21233664UL   // + B*64*36*9
#define OFF_PPV   23592960UL   // + B*64*36
#define OFF_PLC   25952256UL   // + B*64*36
#define OFF_PLV   26017792UL   // + B*64
#define OFF_L2    28377088UL   // + B*64*36
#define OFF_RAW   28377089UL   // + 1
#define OFF_FEAT  30474241UL   // + B*64*32

__device__ __forceinline__ v8f wmma16(v16h a, v16h b, v8f c) {
  return __builtin_amdgcn_wmma_f32_16x16x32_f16(false, a, false, b, (short)0, c, false, false);
}

__device__ __forceinline__ v8f zero8() {
  v8f z;
#pragma unroll
  for (int i = 0; i < 8; ++i) z[i] = 0.f;
  return z;
}

// A fragment (16x32 f16) from row-major f32 global memory, vectorized:
// per lane the needed k's are two contiguous runs of 8 floats.
__device__ __forceinline__ v16h fragA_g(const float* base, int rowstride, int kc, int row0) {
  const int l   = threadIdx.x;
  const int row = row0 + (l & 15);
  const int kb  = kc * 32 + ((l & 16) ? 8 : 0);
  const float* rp = base + (size_t)row * rowstride + kb;
  float4 x0 = ((const float4*)rp)[0];        // k = kb+0..3
  float4 x1 = ((const float4*)rp)[1];        // k = kb+4..7
  float4 y0 = ((const float4*)(rp + 16))[0]; // k = kb+16..19
  float4 y1 = ((const float4*)(rp + 16))[1]; // k = kb+20..23
  v16h a;
  a[0]  = (_Float16)x0.x; a[1]  = (_Float16)x0.y; a[2]  = (_Float16)x0.z; a[3]  = (_Float16)x0.w;
  a[4]  = (_Float16)x1.x; a[5]  = (_Float16)x1.y; a[6]  = (_Float16)x1.z; a[7]  = (_Float16)x1.w;
  a[8]  = (_Float16)y0.x; a[9]  = (_Float16)y0.y; a[10] = (_Float16)y0.z; a[11] = (_Float16)y0.w;
  a[12] = (_Float16)y1.x; a[13] = (_Float16)y1.y; a[14] = (_Float16)y1.z; a[15] = (_Float16)y1.w;
  return a;
}

// A fragment from row-major f16 LDS tile.
__device__ __forceinline__ v16h fragA_lds(const _Float16* base, int stride, int kc, int row0) {
  const int l   = threadIdx.x;
  const int row = row0 + (l & 15);
  const int kb  = kc * 32 + ((l & 16) ? 8 : 0);
  v16h a;
#pragma unroll
  for (int i = 0; i < 16; ++i) {
    int k = kb + (i < 8 ? i : i + 8);
    a[i] = base[row * stride + k];
  }
  return a;
}

// B fragment (32x16 f16) from row-major f32 global weights, full tile.
__device__ __forceinline__ v16h fragB_g(const float* w, int ldn, int kc, int col0) {
  const int l   = threadIdx.x;
  const int col = col0 + (l & 15);
  const int kb  = kc * 32 + ((l & 16) ? 8 : 0);
  v16h b;
#pragma unroll
  for (int i = 0; i < 16; ++i)
    b[i] = (_Float16)w[(size_t)(kb + (i < 8 ? i : i + 8)) * ldn + col];
  return b;
}

// B fragment with column/K guards (edge tiles, short K).
__device__ __forceinline__ v16h fragB_gg(const float* w, int ldn, int kc, int col0,
                                         int ncols, int kmax) {
  const int l   = threadIdx.x;
  const int col = col0 + (l & 15);
  const int kb  = kc * 32 + ((l & 16) ? 8 : 0);
  v16h b;
#pragma unroll
  for (int i = 0; i < 16; ++i) {
    int k = kb + (i < 8 ? i : i + 8);
    float v = (col < ncols && k < kmax) ? w[(size_t)k * ldn + col] : 0.f;
    b[i] = (_Float16)v;
  }
  return b;
}

__device__ __forceinline__ float sigm(float x) { return 1.f / (1.f + __expf(-x)); }

__device__ __forceinline__ void geom(const float p[6], float m[9]) {
  float sx = sigm(p[0]) + 0.01f;
  float sy = sigm(p[1]) + 0.01f;
  float th = p[2] * 6.283185307179586f;
  float sh = tanhf(p[3] * 5.f);
  float tx = tanhf(p[4] * 5.f);
  float ty = tanhf(p[5] * 5.f);
  float s, c;
  __sincosf(th, &s, &c);
  m[0] = sx * c + sh * sy * s;
  m[1] = -sx * s + sh * sy * c;
  m[2] = tx;
  m[3] = sy * s;
  m[4] = sy * c;
  m[5] = ty;
  m[6] = 0.f; m[7] = 0.f; m[8] = 1.f;
}

__global__ void init_out(float* out) {
  if (threadIdx.x == 0 && blockIdx.x == 0) out[OFF_L2] = 0.f;
}

// One wave per block; M = 32 batch rows (two 16-row M-tiles sharing every B fragment).
__global__ __launch_bounds__(32) void capsule_fused(
    const float* __restrict__ feat,
    const float* __restrict__ w1,   const float* __restrict__ b1,
    const float* __restrict__ w2,   const float* __restrict__ b2,
    const float* __restrict__ cmw,  const float* __restrict__ cmb,
    const float* __restrict__ cprw,
    const float* __restrict__ ccrw, const float* __restrict__ ccrb,
    const float* __restrict__ pcw,  const float* __restrict__ pcb,
    const float* __restrict__ pvw,  const float* __restrict__ pvb,
    const float* __restrict__ svw,  const float* __restrict__ svb,
    const float* __restrict__ cstat,
    float* __restrict__ out)
{
  const int n    = blockIdx.y;        // capsule
  const int b0   = blockIdx.x * 32;   // batch tile base (32 rows)
  const int l    = threadIdx.x;
  const int colL = l & 15;
  const int rb   = (l & 16) ? 8 : 0;  // C/D row base within a 16-row M-tile

  __shared__ _Float16 sh_h[32 * DH];      // h tile (f16)
  __shared__ _Float16 sh_cp[32 * 64];     // caps_params padded to K=64
  __shared__ _Float16 sh_cf[32 * DH];     // caps_feat (f16)
  __shared__ float    sh_cprd[32 * 216];  // cpr_dynamic (f32)
  __shared__ float    sh_ccr[32 * 6];
  __shared__ float    sh_pc[32];          // sigmoid(pres_logit_per_caps)

  const float* featb = feat + ((size_t)b0 * NCAPS + n) * DIN;

  // ---- GEMM1: h = relu(x @ w1 + b1)   (32x256)x(256x128) ----
  {
    const float* w1n = w1 + (size_t)n * DIN * DH;
    v8f acc[2][8];
#pragma unroll
    for (int m = 0; m < 2; ++m)
#pragma unroll
      for (int t = 0; t < 8; ++t) acc[m][t] = zero8();
    for (int kc = 0; kc < 8; ++kc) {
      v16h a0 = fragA_g(featb, NCAPS * DIN, kc, 0);
      v16h a1 = fragA_g(featb, NCAPS * DIN, kc, 16);
#pragma unroll
      for (int t = 0; t < 8; ++t) {
        v16h bb = fragB_g(w1n, DH, kc, t * 16);
        acc[0][t] = wmma16(a0, bb, acc[0][t]);
        acc[1][t] = wmma16(a1, bb, acc[1][t]);
      }
    }
#pragma unroll
    for (int t = 0; t < 8; ++t) {
      int col = t * 16 + colL;
      float bias = b1[n * DH + col];
#pragma unroll
      for (int m = 0; m < 2; ++m)
#pragma unroll
        for (int r = 0; r < 8; ++r) {
          float v = acc[m][t][r] + bias;
          sh_h[(m * 16 + r + rb) * DH + col] = (_Float16)(v > 0.f ? v : 0.f);
        }
    }
  }
  __syncthreads();

  // ---- GEMM2: raw = relu(h @ w2 + b2)  (32x128)x(128x32) ----
  {
    const float* w2n = w2 + (size_t)n * DH * DCP;
    v8f acc[2][2];
#pragma unroll
    for (int m = 0; m < 2; ++m)
#pragma unroll
      for (int t = 0; t < 2; ++t) acc[m][t] = zero8();
    for (int kc = 0; kc < 4; ++kc) {
      v16h a0 = fragA_lds(sh_h, DH, kc, 0);
      v16h a1 = fragA_lds(sh_h, DH, kc, 16);
#pragma unroll
      for (int t = 0; t < 2; ++t) {
        v16h bb = fragB_g(w2n, DCP, kc, t * 16);
        acc[0][t] = wmma16(a0, bb, acc[0][t]);
        acc[1][t] = wmma16(a1, bb, acc[1][t]);
      }
    }
    // pad caps_params cols 32..63: [1, 0, 0, ...]
    for (int idx = l; idx < 32 * 32; idx += 32) {
      int row = idx >> 5, c = (idx & 31) + 32;
      sh_cp[row * 64 + c] = (_Float16)((c == 32) ? 1.f : 0.f);
    }
#pragma unroll
    for (int t = 0; t < 2; ++t) {
      int col = t * 16 + colL;
      float bias = b2[n * DCP + col];
#pragma unroll
      for (int m = 0; m < 2; ++m)
#pragma unroll
        for (int r = 0; r < 8; ++r) {
          int row = m * 16 + r + rb;
          float v = acc[m][t][r] + bias;
          v = v > 0.f ? v : 0.f;
          sh_cp[row * 64 + col] = (_Float16)v;
          out[OFF_RAW + ((size_t)(b0 + row) * NCAPS + n) * DCP + col] = v;
        }
    }
  }
  __syncthreads();

  // ---- GEMM3: caps_feat = relu(caps_params @ caps_mlp_w + b)  K=33 ----
  {
    const float* cmwn = cmw + (size_t)n * 33 * DH;
    v8f acc[2][8];
#pragma unroll
    for (int m = 0; m < 2; ++m)
#pragma unroll
      for (int t = 0; t < 8; ++t) acc[m][t] = zero8();
    for (int kc = 0; kc < 2; ++kc) {
      v16h a0 = fragA_lds(sh_cp, 64, kc, 0);
      v16h a1 = fragA_lds(sh_cp, 64, kc, 16);
#pragma unroll
      for (int t = 0; t < 8; ++t) {
        v16h bb = fragB_gg(cmwn, DH, kc, t * 16, DH, 33);
        acc[0][t] = wmma16(a0, bb, acc[0][t]);
        acc[1][t] = wmma16(a1, bb, acc[1][t]);
      }
    }
#pragma unroll
    for (int t = 0; t < 8; ++t) {
      int col = t * 16 + colL;
      float bias = cmb[n * DH + col];
#pragma unroll
      for (int m = 0; m < 2; ++m)
#pragma unroll
        for (int r = 0; r < 8; ++r) {
          float v = acc[m][t][r] + bias;
          sh_cf[(m * 16 + r + rb) * DH + col] = (_Float16)(v > 0.f ? v : 0.f);
        }
    }
  }
  __syncthreads();

  // Preload caps_feat A fragments once; reuse for all five heads.
  v16h af[2][4];
#pragma unroll
  for (int m = 0; m < 2; ++m)
#pragma unroll
    for (int kc = 0; kc < 4; ++kc) af[m][kc] = fragA_lds(sh_cf, DH, kc, m * 16);

  // ---- Head: cpr_dynamic (128 -> 216), no bias ----
  {
    const float* wn = cprw + (size_t)n * DH * 216;
    for (int t = 0; t < 14; ++t) {
      v8f a0 = zero8(), a1 = zero8();
#pragma unroll
      for (int kc = 0; kc < 4; ++kc) {
        v16h bb = fragB_gg(wn, 216, kc, t * 16, 216, DH);
        a0 = wmma16(af[0][kc], bb, a0);
        a1 = wmma16(af[1][kc], bb, a1);
      }
      int col = t * 16 + colL;
      if (col < 216) {
#pragma unroll
        for (int r = 0; r < 8; ++r) {
          sh_cprd[(r + rb) * 216 + col]      = a0[r];
          sh_cprd[(16 + r + rb) * 216 + col] = a1[r];
        }
      }
    }
  }

  // ---- Head: ccr (128 -> 6) ----
  {
    const float* wn = ccrw + (size_t)n * DH * 6;
    v8f a0 = zero8(), a1 = zero8();
#pragma unroll
    for (int kc = 0; kc < 4; ++kc) {
      v16h bb = fragB_gg(wn, 6, kc, 0, 6, DH);
      a0 = wmma16(af[0][kc], bb, a0);
      a1 = wmma16(af[1][kc], bb, a1);
    }
    if (colL < 6) {
      float bias = ccrb[n * 6 + colL];
#pragma unroll
      for (int r = 0; r < 8; ++r) {
        sh_ccr[(r + rb) * 6 + colL]      = a0[r] + bias;
        sh_ccr[(16 + r + rb) * 6 + colL] = a1[r] + bias;
      }
    }
  }

  // ---- Head: pres_logit_per_caps (128 -> 1) ----
  {
    const float* wn = pcw + (size_t)n * DH;
    v8f a0 = zero8(), a1 = zero8();
#pragma unroll
    for (int kc = 0; kc < 4; ++kc) {
      v16h bb = fragB_gg(wn, 1, kc, 0, 1, DH);
      a0 = wmma16(af[0][kc], bb, a0);
      a1 = wmma16(af[1][kc], bb, a1);
    }
    if (colL == 0) {
      float bias = pcb[n];
#pragma unroll
      for (int r = 0; r < 8; ++r) {
        int r0 = r + rb, r1 = 16 + r + rb;
        float lg0 = a0[r] + bias, lg1 = a1[r] + bias;
        out[OFF_PLC + (size_t)(b0 + r0) * NCAPS + n] = lg0;
        out[OFF_PLC + (size_t)(b0 + r1) * NCAPS + n] = lg1;
        sh_pc[r0] = sigm(lg0);
        sh_pc[r1] = sigm(lg1);
      }
    }
  }
  __syncthreads();

  // ---- Head: pres_logit_per_vote (128 -> 36) + pres_per_vote ----
  {
    const float* wn = pvw + (size_t)n * DH * NVOTES;
    for (int t = 0; t < 3; ++t) {
      v8f a0 = zero8(), a1 = zero8();
#pragma unroll
      for (int kc = 0; kc < 4; ++kc) {
        v16h bb = fragB_gg(wn, NVOTES, kc, t * 16, NVOTES, DH);
        a0 = wmma16(af[0][kc], bb, a0);
        a1 = wmma16(af[1][kc], bb, a1);
      }
      int col = t * 16 + colL;
      if (col < NVOTES) {
        float bias = pvb[n * NVOTES + col];
#pragma unroll
        for (int m = 0; m < 2; ++m)
#pragma unroll
          for (int r = 0; r < 8; ++r) {
            int row = m * 16 + r + rb;
            float logit = (m == 0 ? a0[r] : a1[r]) + bias;
            size_t idx = ((size_t)(b0 + row) * NCAPS + n) * NVOTES + col;
            out[OFF_PLV + idx] = logit;
            out[OFF_PPV + idx] = sh_pc[row] * sigm(logit);
          }
      }
    }
  }

  // ---- Head: scale_per_vote (128 -> 36), softplus(x+0.5)+0.01 ----
  {
    const float* wn = svw + (size_t)n * DH * NVOTES;
    for (int t = 0; t < 3; ++t) {
      v8f a0 = zero8(), a1 = zero8();
#pragma unroll
      for (int kc = 0; kc < 4; ++kc) {
        v16h bb = fragB_gg(wn, NVOTES, kc, t * 16, NVOTES, DH);
        a0 = wmma16(af[0][kc], bb, a0);
        a1 = wmma16(af[1][kc], bb, a1);
      }
      int col = t * 16 + colL;
      if (col < NVOTES) {
        float bias = svb[n * NVOTES + col];
#pragma unroll
        for (int m = 0; m < 2; ++m)
#pragma unroll
          for (int r = 0; r < 8; ++r) {
            int row = m * 16 + r + rb;
            float x = (m == 0 ? a0[r] : a1[r]) + bias + 0.5f;
            float sp = (x > 20.f) ? x : log1pf(__expf(x));
            out[OFF_SCALE + ((size_t)(b0 + row) * NCAPS + n) * NVOTES + col] = sp + 0.01f;
          }
      }
    }
  }

  // ---- Votes: geometric transforms + 3x3 matmul; L2 of cpr_dynamic ----
  float l2 = 0.f;
  for (int idx = l; idx < 32 * NVOTES; idx += 32) {
    int row = idx / NVOTES;
    int v   = idx - row * NVOTES;
    float pd[6], pc6[6];
#pragma unroll
    for (int j = 0; j < 6; ++j) {
      float d = sh_cprd[row * 216 + v * 6 + j];
      l2 += d * d;
      pd[j] = d + cstat[((size_t)n * NVOTES + v) * 6 + j];
    }
#pragma unroll
    for (int j = 0; j < 6; ++j) pc6[j] = sh_ccr[row * 6 + j];
    float A[9], Bm[9];
    geom(pc6, A);
    geom(pd, Bm);
    size_t base = OFF_VOTES + (((size_t)(b0 + row) * NCAPS + n) * NVOTES + v) * 9;
#pragma unroll
    for (int i = 0; i < 3; ++i)
#pragma unroll
      for (int j = 0; j < 3; ++j)
        out[base + i * 3 + j] =
            A[i * 3] * Bm[j] + A[i * 3 + 1] * Bm[3 + j] + A[i * 3 + 2] * Bm[6 + j];
  }
#pragma unroll
  for (int off = 16; off > 0; off >>= 1) l2 += __shfl_down(l2, off, 32);
  if (l == 0) atomicAdd(out + OFF_L2, l2 * (1.f / 2048.f));

  // ---- features passthrough (float4 both sides; OFF_FEAT is odd, so out side is not 16B aligned) ----
  {
    const float4* src = (const float4*)featb;
    for (int idx = l; idx < 32 * (DIN / 4); idx += 32) {
      int row = idx >> 6, q = idx & 63;  // 64 float4 per row
      float4 val = src[(size_t)row * (NCAPS * DIN / 4) + q];
      ((float4*)(out + OFF_FEAT))[((size_t)(b0 + row) * NCAPS + n) * (DIN / 4) + q] = val;
    }
  }
}

extern "C" void kernel_launch(void* const* d_in, const int* in_sizes, int n_in,
                              void* d_out, int out_size, void* d_ws, size_t ws_size,
                              hipStream_t stream) {
  (void)in_sizes; (void)n_in; (void)out_size; (void)d_ws; (void)ws_size;
  const float* feat  = (const float*)d_in[0];
  const float* w1    = (const float*)d_in[1];
  const float* b1    = (const float*)d_in[2];
  const float* w2    = (const float*)d_in[3];
  const float* b2    = (const float*)d_in[4];
  const float* cmw   = (const float*)d_in[5];
  const float* cmb   = (const float*)d_in[6];
  const float* cprw  = (const float*)d_in[7];
  const float* ccrw  = (const float*)d_in[8];
  const float* ccrb  = (const float*)d_in[9];
  const float* pcw   = (const float*)d_in[10];
  const float* pcb   = (const float*)d_in[11];
  const float* pvw   = (const float*)d_in[12];
  const float* pvb   = (const float*)d_in[13];
  const float* svw   = (const float*)d_in[14];
  const float* svb   = (const float*)d_in[15];
  const float* cstat = (const float*)d_in[16];
  float* out = (float*)d_out;

  init_out<<<1, 32, 0, stream>>>(out);
  dim3 grid(32, NCAPS);  // 1024/32 batch tiles x 64 capsules
  capsule_fused<<<grid, 32, 0, stream>>>(feat, w1, b1, w2, b2, cmw, cmb, cprw,
                                         ccrw, ccrb, pcw, pcb, pvw, pvb, svw, svb,
                                         cstat, out);
}
